// TransformerLayerController_49881750176073
// MI455X (gfx1250) — matrix operand order repaired and checked by reading, not compiled, not run on hardware
//
#include <hip/hip_runtime.h>

typedef __attribute__((ext_vector_type(16))) __bf16        v16bf;
typedef __attribute__((ext_vector_type(8)))  float         v8f;
typedef __attribute__((ext_vector_type(8)))  unsigned int  v8u;

union FragU { v8u u; v16bf b; };

constexpr int H  = 16;
constexpr int S  = 2048;
constexpr int D  = 128;
constexpr int QT = 128;       // query rows per block (8 waves x 16)
constexpr int KT = 32;        // key rows per tile (fp32 staging double buffered)
constexpr int LDKF = D + 4;   // float stride for fp32 K/V staging rows
constexpr int LDKB = D + 8;   // ushort stride for bf16 K tile rows (272 B, 16B-aligned)
constexpr int LDV  = KT + 8;  // ushort stride for transposed bf16 V tile rows (80 B)
constexpr int LDP  = KT + 8;  // ushort stride for P staging rows
constexpr float SM_SCALE = 0.088388347648318447f;  // 1/sqrt(128)

__device__ inline unsigned int pack_bf16x2(float a, float b) {
  unsigned int ua = __float_as_uint(a);
  unsigned int ub = __float_as_uint(b);
  ua += 0x7FFFu + ((ua >> 16) & 1u);   // round-to-nearest-even
  ub += 0x7FFFu + ((ub >> 16) & 1u);
  return (ua >> 16) | (ub & 0xFFFF0000u);
}

__device__ inline unsigned short f2bf(float a) {
  unsigned int u = __float_as_uint(a);
  u += 0x7FFFu + ((u >> 16) & 1u);
  return (unsigned short)(u >> 16);
}

// ---------------------------------------------------------------------------
// Flash attention: async global->LDS DMA (double buffered, ASYNCcnt), one
// cooperative fp32->bf16 conversion per tile, bf16 WMMA inner loops.
// Grid: (S/QT, H), block 256 (8 wave32 waves).
// ---------------------------------------------------------------------------
__global__ __launch_bounds__(256) void attn_kernel(const float* __restrict__ Q,
                                                   const float* __restrict__ K,
                                                   const float* __restrict__ V,
                                                   float* __restrict__ out) {
  __shared__ float Ksf[2][KT * LDKF];        // raw fp32 async staging
  __shared__ float Vsf[2][KT * LDKF];
  __shared__ unsigned short Kbf[KT * LDKB];  // bf16, row-major [key][d]
  __shared__ unsigned short Vt[D * LDV];     // bf16, transposed [d][key]
  __shared__ unsigned short Ps[8 * 16 * LDP];

  const int tid  = threadIdx.x;
  const int lane = tid & 31;
  const int wave = tid >> 5;
  const int half = lane >> 4;
  const int l16  = lane & 15;
  const int h    = blockIdx.y;
  const int qBase = blockIdx.x * QT;
  const int qRowBase = qBase + wave * 16;

  // Issue one K/V tile's async global->LDS copies (ASYNCcnt-tracked DMA).
  auto issue_tile = [&](int buf, int kBase) {
    for (int i = tid; i < KT * (D / 4); i += 256) {
      const int row = i >> 5;            // 32 float4 chunks per row
      const int c4  = (i & 31) << 2;
      const float* gk = K + ((size_t)h * S + kBase + row) * D + c4;
      const float* gv = V + ((size_t)h * S + kBase + row) * D + c4;
      unsigned int lk = (unsigned int)(size_t)&Ksf[buf][row * LDKF + c4];
      unsigned int lv = (unsigned int)(size_t)&Vsf[buf][row * LDKF + c4];
      asm volatile("global_load_async_to_lds_b128 %0, %1, off"
                   :: "v"(lk), "v"(gk) : "memory");
      asm volatile("global_load_async_to_lds_b128 %0, %1, off"
                   :: "v"(lv), "v"(gv) : "memory");
    }
  };

  // Preload this wave's Q slab straight into WMMA A-matrix layout (bf16).
  FragU qf[4];
  {
    const float* qp = Q + ((size_t)h * S + (qRowBase + l16)) * D;
#pragma unroll
    for (int kc = 0; kc < 4; ++kc) {
#pragma unroll
      for (int v = 0; v < 8; ++v) {
        const int koff = (v < 4 ? 2 * v : 16 + 2 * (v - 4)) + (half ? 8 : 0);
        const int d0 = kc * 32 + koff;
        qf[kc].u[v] = pack_bf16x2(qp[d0], qp[d0 + 1]);
      }
    }
  }

  float m_i[8], l_i[8];
  v8f O[8];
#pragma unroll
  for (int r = 0; r < 8; ++r) { m_i[r] = -1e30f; l_i[r] = 0.0f; }
#pragma unroll
  for (int dg = 0; dg < 8; ++dg) O[dg] = {};

  unsigned short* Pw = &Ps[wave * 16 * LDP];
  const int ktiles = qBase / KT + 4;

  // Prologue: start DMA for tile 0.
  issue_tile(0, 0);

  for (int kt = 0; kt < ktiles; ++kt) {
    const int cur = kt & 1;
    const int kBase = kt * KT;

    // Tile kt's fp32 staging has landed; previous tile's compute is done.
    asm volatile("s_wait_asynccnt 0x0" ::: "memory");
    __syncthreads();

    // Kick off next tile's DMA; it runs under convert + WMMAs below.
    if (kt + 1 < ktiles) issue_tile(cur ^ 1, kBase + KT);

    // One cooperative fp32 -> bf16 conversion per tile (no per-wave redo).
    for (int i = tid; i < KT * (D / 2); i += 256) {   // K: row-major pairs
      const int row = i >> 6;
      const int c2  = (i & 63) << 1;
      const float* src = &Ksf[cur][row * LDKF + c2];
      *(unsigned int*)&Kbf[row * LDKB + c2] = pack_bf16x2(src[0], src[1]);
    }
    for (int i = tid; i < D * (KT / 2); i += 256) {   // V: transpose to [d][key]
      const int d  = i >> 4;
      const int k2 = (i & 15) << 1;
      *(unsigned int*)&Vt[d * LDV + k2] =
          pack_bf16x2(Vsf[cur][k2 * LDKF + d], Vsf[cur][(k2 + 1) * LDKF + d]);
    }
    __syncthreads();

    // Scores: 16 q rows x 32 keys per wave, D reduced by 4 WMMAs of K=32.
    float sc[2][8];
#pragma unroll
    for (int cg = 0; cg < 2; ++cg) {
      v8f acc = {};
#pragma unroll
      for (int kc = 0; kc < 4; ++kc) {
        FragU Bk;
        const unsigned short* kr = &Kbf[(cg * 16 + l16) * LDKB + kc * 32];
#pragma unroll
        for (int v = 0; v < 8; ++v)
          Bk.u[v] = *(const unsigned int*)&kr[(v < 4 ? 2 * v : 16 + 2 * (v - 4)) + (half ? 8 : 0)];
        acc = __builtin_amdgcn_wmma_f32_16x16x32_bf16(false, qf[kc].b, false, Bk.b,
                                                      (short)0, acc, false, false);
      }
      const int key = kBase + cg * 16 + l16;
#pragma unroll
      for (int r = 0; r < 8; ++r) {
        const int qrow = qRowBase + r + half * 8;   // C layout: rows r / r+8 by half
        sc[cg][r] = (key <= qrow) ? acc[r] * SM_SCALE : -1e30f;
      }
    }

    // Online softmax (rows live across 16-lane half-groups -> shfl_xor reduce).
#pragma unroll
    for (int r = 0; r < 8; ++r) {
      float mx = fmaxf(sc[0][r], sc[1][r]);
#pragma unroll
      for (int off = 1; off < 16; off <<= 1)
        mx = fmaxf(mx, __shfl_xor(mx, off, 32));
      const float mnew  = fmaxf(m_i[r], mx);
      const float alpha = __expf(m_i[r] - mnew);
      m_i[r] = mnew;
      float psum = 0.0f;
#pragma unroll
      for (int cg = 0; cg < 2; ++cg) {
        const float p = __expf(sc[cg][r] - mnew);
        sc[cg][r] = p;
        psum += p;
      }
#pragma unroll
      for (int off = 1; off < 16; off <<= 1)
        psum += __shfl_xor(psum, off, 32);
      l_i[r] = l_i[r] * alpha + psum;
#pragma unroll
      for (int dg = 0; dg < 8; ++dg) O[dg][r] *= alpha;
    }

    // Stage P (bf16) to per-wave LDS region so it can re-enter A-layout.
#pragma unroll
    for (int cg = 0; cg < 2; ++cg)
#pragma unroll
      for (int r = 0; r < 8; ++r)
        Pw[(r + half * 8) * LDP + cg * 16 + l16] = f2bf(sc[cg][r]);
    __syncthreads();

    // O += P * V  (one K=32 chunk, 8 d-column groups; V is pre-transposed).
    {
      FragU Af;
      const unsigned short* pr = &Pw[l16 * LDP];
#pragma unroll
      for (int v = 0; v < 8; ++v) {
        const int koff = (v < 4 ? 2 * v : 16 + 2 * (v - 4)) + (half ? 8 : 0);
        Af.u[v] = *(const unsigned int*)&pr[koff];
      }
#pragma unroll
      for (int dg = 0; dg < 8; ++dg) {
        FragU Bv;
        const unsigned short* vr = &Vt[(dg * 16 + l16) * LDV];
#pragma unroll
        for (int v = 0; v < 8; ++v)
          Bv.u[v] = *(const unsigned int*)&vr[(v < 4 ? 2 * v : 16 + 2 * (v - 4)) + (half ? 8 : 0)];
        O[dg] = __builtin_amdgcn_wmma_f32_16x16x32_bf16(false, Af.b, false, Bv.b,
                                                        (short)0, O[dg], false, false);
      }
    }
  }

  // Normalize and store.
#pragma unroll
  for (int dg = 0; dg < 8; ++dg) {
#pragma unroll
    for (int r = 0; r < 8; ++r) {
      const int qrow = qRowBase + r + half * 8;
      out[((size_t)h * S + qrow) * D + dg * 16 + l16] = O[dg][r] / l_i[r];
    }
  }
}

// ---------------------------------------------------------------------------
// K quantization: one block per (h,d) channel slice of S tokens.
// ---------------------------------------------------------------------------
__global__ __launch_bounds__(256) void kquant_kernel(const float* __restrict__ K,
    float* __restrict__ kq, float* __restrict__ kscale,
    float* __restrict__ kspv, float* __restrict__ kspi) {
  __shared__ float buf[S];
  __shared__ float red[256];
  __shared__ int  redi[256];
  const int tid = threadIdx.x;
  const int bx = blockIdx.x;            // 0..H*D-1
  const int h = bx >> 7, d = bx & 127;

  for (int i = tid; i < S; i += 256) {
    const float x = K[((size_t)h * S + i) * D + d];
    buf[i] = (i < 4) ? 0.0f : x;        // sink tokens zeroed before isolation
  }
  __syncthreads();

  for (int j = 0; j < 8; ++j) {         // top-8 |.| tokens, lowest-index ties
    float bm = -1.0f; int bi = S;
    for (int i = tid; i < S; i += 256) {
      const float a = fabsf(buf[i]);
      if (a > bm) { bm = a; bi = i; }
    }
    red[tid] = bm; redi[tid] = bi;
    __syncthreads();
    for (int s2 = 128; s2 > 0; s2 >>= 1) {
      if (tid < s2) {
        const float om = red[tid + s2]; const int oi = redi[tid + s2];
        if (om > red[tid] || (om == red[tid] && oi < redi[tid])) {
          red[tid] = om; redi[tid] = oi;
        }
      }
      __syncthreads();
    }
    const int idx = redi[0];
    if (tid == 0) {
      kspv[bx * 8 + j] = buf[idx];
      kspi[bx * 8 + j] = (float)((h * S + idx) * D + d);
      buf[idx] = 0.0f;
    }
    __syncthreads();
  }

  float am = 0.0f;
  for (int i = tid; i < S; i += 256) am = fmaxf(am, fabsf(buf[i]));
  red[tid] = am;
  __syncthreads();
  for (int s2 = 128; s2 > 0; s2 >>= 1) {
    if (tid < s2) red[tid] = fmaxf(red[tid], red[tid + s2]);
    __syncthreads();
  }
  const float scale = fmaxf(red[0], 1e-8f) / 7.0f;
  if (tid == 0) kscale[bx] = scale;
  for (int i = tid; i < S; i += 256) {
    float q = rintf(buf[i] / scale);
    q = fminf(7.0f, fmaxf(-7.0f, q));
    kq[((size_t)h * S + i) * D + d] = q;
  }
  if (tid < 4) {                        // full-precision sink sparse entries
    kspv[16384 + (h * 4 + tid) * 128 + d] = K[((size_t)h * S + tid) * D + d];
    kspi[16384 + (h * 4 + tid) * 128 + d] = (float)((h * S + tid) * D + d);
  }
}

// ---------------------------------------------------------------------------
// V quantization: one wave per (h,s) token slice of D channels.
// ---------------------------------------------------------------------------
__global__ __launch_bounds__(256) void vquant_kernel(const float* __restrict__ V,
    float* __restrict__ vq, float* __restrict__ vscale,
    float* __restrict__ vspv, float* __restrict__ vspi) {
  const int lane = threadIdx.x & 31;
  const int wave = threadIdx.x >> 5;
  const int t = blockIdx.x * 8 + wave;  // token index 0..H*S-1
  const int h = t >> 11, s = t & 2047;

  const float4 o4 = *(const float4*)(V + (size_t)t * D + lane * 4);
  float orig[4] = {o4.x, o4.y, o4.z, o4.w};
  float x[4], a[4];
#pragma unroll
  for (int c = 0; c < 4; ++c) {
    x[c] = (s < 4) ? 0.0f : orig[c];
    a[c] = fabsf(x[c]);
  }

#pragma unroll
  for (int j = 0; j < 4; ++j) {         // top-4 |.| channels
    float bm = -1.0f; int bi = 0x7fffffff; float bv = 0.0f;
#pragma unroll
    for (int c = 0; c < 4; ++c)
      if (a[c] > bm) { bm = a[c]; bi = lane * 4 + c; bv = x[c]; }
#pragma unroll
    for (int off = 16; off > 0; off >>= 1) {
      const float om = __shfl_xor(bm, off, 32);
      const int   oi = __shfl_xor(bi, off, 32);
      const float ov = __shfl_xor(bv, off, 32);
      if (om > bm || (om == bm && oi < bi)) { bm = om; bi = oi; bv = ov; }
    }
    if (lane == 0) {
      vspv[t * 4 + j] = bv;
      vspi[t * 4 + j] = (float)(t * D + bi);
    }
    if ((bi >> 2) == lane) { x[bi & 3] = 0.0f; a[bi & 3] = -1.0f; }
  }

  float am = 0.0f;
#pragma unroll
  for (int c = 0; c < 4; ++c) am = fmaxf(am, fabsf(x[c]));
#pragma unroll
  for (int off = 16; off > 0; off >>= 1) am = fmaxf(am, __shfl_xor(am, off, 32));
  const float scale = fmaxf(am, 1e-8f) / 7.0f;
  if (lane == 0) vscale[t] = scale;

  float qv[4];
#pragma unroll
  for (int c = 0; c < 4; ++c) {
    float q = rintf(x[c] / scale);
    qv[c] = fminf(7.0f, fmaxf(-7.0f, q));
  }
  *(float4*)(vq + (size_t)t * D + lane * 4) = make_float4(qv[0], qv[1], qv[2], qv[3]);

  if (s < 4) {                          // full-precision sink sparse entries
#pragma unroll
    for (int c = 0; c < 4; ++c) {
      const int d = lane * 4 + c;
      vspv[131072 + (h * 4 + s) * 128 + d] = orig[c];
      vspi[131072 + (h * 4 + s) * 128 + d] = (float)(t * D + d);
    }
  }
}

extern "C" void kernel_launch(void* const* d_in, const int* in_sizes, int n_in,
                              void* d_out, int out_size, void* d_ws, size_t ws_size,
                              hipStream_t stream) {
  const float* q = (const float*)d_in[0];
  const float* k = (const float*)d_in[1];
  const float* v = (const float*)d_in[2];
  float* out = (float*)d_out;

  const size_t N = (size_t)H * S * D;   // 4,194,304 (B=1)
  float* attn   = out;
  float* kq     = out + N;
  float* kscale = out + 2 * N;          // H*D = 2048
  float* kspv   = kscale + 2048;        // 16384 + 8192 = 24576
  float* kspi   = kspv + 24576;
  float* vq     = kspi + 24576;
  float* vscale = vq + N;               // H*S = 32768
  float* vspv   = vscale + 32768;       // 131072 + 8192 = 139264
  float* vspi   = vspv + 139264;

  dim3 ag(S / QT, H);
  attn_kernel<<<ag, 256, 0, stream>>>(q, k, v, attn);
  kquant_kernel<<<H * D, 256, 0, stream>>>(k, kq, kscale, kspv, kspi);
  vquant_kernel<<<(H * S) / 8, 256, 0, stream>>>(v, vq, vscale, vspv, vspi);
}
